// HyperbolicGATLayer_33818572489036
// MI455X (gfx1250) — compile-verified, matrix operand order checked by reading, not run on hardware
//
#include <hip/hip_runtime.h>
#include <hip/hip_bf16.h>
#include <cmath>

typedef _Float16 v8h  __attribute__((ext_vector_type(8)));
typedef _Float16 v16h __attribute__((ext_vector_type(16)));
typedef float    v8f  __attribute__((ext_vector_type(8)));

#define NN    4096
#define INC   256
#define OUTC  256
#define HEADS 4
#define DH    64

// Load a 16x32 f16 WMMA fragment half for this lane.
// Caller passes p = rowBase + khalf where khalf = (lane<16 ? 0 : 8);
// fragment halves are elements [0..7] and [16..23] relative to p.
__device__ __forceinline__ v16h load_frag(const _Float16* p) {
    v8h lo = *(const v8h*)(p);
    v8h hi = *(const v8h*)(p + 16);
    return __builtin_shufflevector(lo, hi, 0,1,2,3,4,5,6,7,8,9,10,11,12,13,14,15);
}

// ---- CDNA5 async global->LDS (ASYNCcnt-tracked) helpers ------------------
// LDS operand = low 32 bits of the generic pointer (LDS aperture offset).
__device__ __forceinline__ unsigned lds_off(const void* p) {
    return (unsigned)(uintptr_t)p;
}
__device__ __forceinline__ void async_load_b128(unsigned ldsaddr, const void* gaddr) {
    asm volatile("global_load_async_to_lds_b128 %0, %1, off"
                 :: "v"(ldsaddr), "v"(gaddr) : "memory");
}
__device__ __forceinline__ void async_load_b32(unsigned ldsaddr, const void* gaddr) {
    asm volatile("global_load_async_to_lds_b32 %0, %1, off"
                 :: "v"(ldsaddr), "v"(gaddr) : "memory");
}

// ---------------- Kernel 1: logmap0 + f16 convert -------------------------
// One wave per row: norm over 256 elems, scale = artanh(min(n,MAX))/n.
__global__ void logmap_kernel(const float* __restrict__ x, _Float16* __restrict__ xh) {
    const int lane = threadIdx.x & 31;
    const int wave = threadIdx.x >> 5;
    const int row  = blockIdx.x * 8 + wave;
    const float* xr = x + (size_t)row * INC;
    float4 v0 = ((const float4*)xr)[lane * 2 + 0];
    float4 v1 = ((const float4*)xr)[lane * 2 + 1];
    float ss = v0.x*v0.x + v0.y*v0.y + v0.z*v0.z + v0.w*v0.w
             + v1.x*v1.x + v1.y*v1.y + v1.z*v1.z + v1.w*v1.w;
    #pragma unroll
    for (int off = 16; off > 0; off >>= 1) ss += __shfl_xor(ss, off, 32);
    float n   = sqrtf(ss);
    float xn  = fmaxf(n, 1e-15f);
    float arg = fminf(xn, 1.0f - 1e-5f);
    float scale = 0.5f * __logf((1.0f + arg) / (1.0f - arg)) / xn;
    _Float16* o = xh + (size_t)row * INC + lane * 8;
    v8h r;
    r[0] = (_Float16)(scale * v0.x); r[1] = (_Float16)(scale * v0.y);
    r[2] = (_Float16)(scale * v0.z); r[3] = (_Float16)(scale * v0.w);
    r[4] = (_Float16)(scale * v1.x); r[5] = (_Float16)(scale * v1.y);
    r[6] = (_Float16)(scale * v1.z); r[7] = (_Float16)(scale * v1.w);
    *(v8h*)o = r;
}

// ---------------- Kernel 2: W -> f16 --------------------------------------
__global__ void wcvt_kernel(const float* __restrict__ W, _Float16* __restrict__ Wh) {
    int t = blockIdx.x * blockDim.x + threadIdx.x; // 65536 threads
    Wh[t] = (_Float16)W[t];
}

// ---------------- Kernel 3: h = x_tan @ W^T, stored transposed ------------
// Block = 128 threads (4 waves). Block b -> 16 rows of n; wave w -> 64 cols of o.
// Output hT[o, n] (o-major) so the attention kernel's B loads are contiguous in j.
__global__ void gemm_h_kernel(const _Float16* __restrict__ xh,
                              const _Float16* __restrict__ Wh,
                              _Float16* __restrict__ hT) {
    const int lane  = threadIdx.x & 31;
    const int wave  = threadIdx.x >> 5;
    const int n0    = blockIdx.x * 16;
    const int o0    = wave * 64;
    const int idx   = lane & 15;
    const int khalf = (lane < 16) ? 0 : 8;
    v8f acc[4] = {};
    for (int k0 = 0; k0 < INC; k0 += 32) {
        v16h a = load_frag(xh + (size_t)(n0 + idx) * INC + k0 + khalf);
        #pragma unroll
        for (int t = 0; t < 4; ++t) {
            // B[k][o]: o = o0+t*16+idx fixed per lane; W stored (o,k) row-major -> contiguous
            v16h b = load_frag(Wh + (size_t)(o0 + t * 16 + idx) * INC + k0 + khalf);
            acc[t] = __builtin_amdgcn_wmma_f32_16x16x32_f16(
                false, a, false, b, (short)0, acc[t], false, false);
        }
    }
    const int hi8 = (lane < 16) ? 0 : 8;
    #pragma unroll
    for (int t = 0; t < 4; ++t) {
        #pragma unroll
        for (int v = 0; v < 8; ++v) {
            int n = n0 + v + hi8;
            int o = o0 + t * 16 + idx;
            hT[(size_t)o * NN + n] = (_Float16)acc[t][v];
        }
    }
}

// ---------------- Kernel 4: e_l, e_r --------------------------------------
// Thread i reads column i of hT (coalesced across threads).
__global__ void e_kernel(const _Float16* __restrict__ hT,
                         const float* __restrict__ a_l, const float* __restrict__ a_r,
                         float* __restrict__ e_l, float* __restrict__ e_r) {
    const int i = blockIdx.x * blockDim.x + threadIdx.x; // 0..4095
    #pragma unroll
    for (int h = 0; h < HEADS; ++h) {
        float el = 0.f, er = 0.f;
        for (int d = 0; d < DH; ++d) {
            int r = h * DH + d;
            float v = (float)hT[(size_t)r * NN + i];
            el += v * a_l[r];   // a_l flat (H,D): index h*64+d == r
            er += v * a_r[r];
        }
        e_l[i * HEADS + h] = el;
        e_r[i * HEADS + h] = er;
    }
}

// ---------------- Kernel 5: fused attention -------------------------------
// Block = 128 threads (4 waves), one i-tile of 16 rows; wave = head.
// Single pass over j with double-buffered ASYNC global->LDS staging of the
// adj tile + e_r tile: the next tile streams into LDS while the current one
// feeds the P-build (exp/leaky/mask) and 4 WMMAs per wave. Epilogue divides
// by the row-sum, adds bias, SiLU, cross-head row norm (LDS atomics), expmap0.
__global__ void attn_kernel(const float* __restrict__ adj,
                            const _Float16* __restrict__ hT,
                            const float* __restrict__ e_l,
                            const float* __restrict__ e_r,
                            const float* __restrict__ bias,
                            float* __restrict__ out) {
    __shared__ float adj_s[2][16 * 32];
    __shared__ float er_s[2][32 * HEADS];
    __shared__ float rs_s[HEADS * 16];
    __shared__ float ss_s[16];

    const int tid   = threadIdx.x;
    const int lane  = tid & 31;
    const int head  = tid >> 5;        // wave id == head
    const int idx   = lane & 15;       // A row / C col
    const int khalf = (lane < 16) ? 0 : 8;
    const int hi8   = (lane < 16) ? 0 : 8;
    const int i0    = blockIdx.x * 16;

    // per-thread staging coordinates: 128 threads cover 16x32 floats as b128
    const int sr = tid >> 3;          // 0..15 (i row)
    const int sc = (tid & 7) * 4;     // 0,4,...,28 (j col)

    if (tid < 16) ss_s[tid] = 0.0f;

    const float el = e_l[(size_t)(i0 + idx) * HEADS + head];
    v8f acc[4] = {};
    float rs = 0.0f;

    const int NT = NN / 32;
    // prologue: async-stage tile 0 into buffer 0
    async_load_b128(lds_off(&adj_s[0][sr * 32 + sc]),
                    adj + (size_t)(i0 + sr) * NN + sc);
    async_load_b32(lds_off(&er_s[0][tid]), e_r + tid);

    for (int jt = 0; jt < NT; ++jt) {
        const int cur = jt & 1;
        const int j0  = jt * 32;
        // all waves finished reading buf[1-cur] (iteration jt-1) before refill
        __syncthreads();
        if (jt + 1 < NT) {
            const int nj0 = j0 + 32;
            async_load_b128(lds_off(&adj_s[1 - cur][sr * 32 + sc]),
                            adj + (size_t)(i0 + sr) * NN + nj0 + sc);
            async_load_b32(lds_off(&er_s[1 - cur][tid]),
                           e_r + (size_t)nj0 * HEADS + tid);
            // wait for the *current* tile's 2 older async ops; next tile stays in flight
            asm volatile("s_wait_asynccnt 0x2" ::: "memory");
        } else {
            asm volatile("s_wait_asynccnt 0x0" ::: "memory");
        }
        __syncthreads();   // current tile fully resident for every wave

        // Build P fragment: w = mask ? exp(leaky(e_l+e_r)) : 0 (no max-sub
        // needed: logits are small; self-loop guarantees rowsum > 0).
        _Float16 pa[16];
        #pragma unroll
        for (int g = 0; g < 2; ++g) {
            #pragma unroll
            for (int kk = 0; kk < 8; ++kk) {
                int k = g * 16 + khalf + kk;
                float a  = adj_s[cur][idx * 32 + k];
                float lg = el + er_s[cur][k * HEADS + head];
                lg = (lg > 0.0f) ? lg : 0.2f * lg;
                bool msk = (a > 0.0f) || ((i0 + idx) == (j0 + k));
                float w = msk ? __expf(lg) : 0.0f;
                rs += w;
                pa[g * 8 + kk] = (_Float16)w;
            }
        }
        v16h A;
        #pragma unroll
        for (int q = 0; q < 16; ++q) A[q] = pa[q];

        #pragma unroll
        for (int t = 0; t < 4; ++t) {
            // B[k=j][n=d]: hT row = head*64 + t*16 + idx, contiguous in j
            v16h B = load_frag(hT + (size_t)(head * DH + t * 16 + idx) * NN + j0 + khalf);
            acc[t] = __builtin_amdgcn_wmma_f32_16x16x32_f16(
                false, A, false, B, (short)0, acc[t], false, false);
        }
    }

    // combine the two K-half rowsum partials (lane L and L^16 share row idx)
    rs += __shfl_xor(rs, 16, 32);
    if (lane < 16) rs_s[head * 16 + lane] = rs;
    __syncthreads();

    float inv[8];
    #pragma unroll
    for (int v = 0; v < 8; ++v) inv[v] = 1.0f / rs_s[head * 16 + v + hi8];

    // divide by rowsum, + bias, SiLU; accumulate per-row sum of squares
    float sq[8] = {};
    #pragma unroll
    for (int t = 0; t < 4; ++t) {
        float b = bias[head * DH + t * 16 + idx];
        #pragma unroll
        for (int v = 0; v < 8; ++v) {
            float val = acc[t][v] * inv[v] + b;
            float s   = val / (1.0f + __expf(-val));   // silu
            acc[t][v] = s;
            sq[v] += s * s;
        }
    }
    // reduce sumsq over the 16 lanes of each half, combine heads via LDS
    #pragma unroll
    for (int v = 0; v < 8; ++v) {
        float s = sq[v];
        s += __shfl_xor(s, 1, 32);
        s += __shfl_xor(s, 2, 32);
        s += __shfl_xor(s, 4, 32);
        s += __shfl_xor(s, 8, 32);
        if (idx == 0) atomicAdd(&ss_s[v + hi8], s);
    }
    __syncthreads();

    // expmap0: scale = tanh(||row||)/||row||
    float scale[8];
    #pragma unroll
    for (int v = 0; v < 8; ++v) {
        float nrm = sqrtf(ss_s[v + hi8]);
        float vn  = fmaxf(nrm, 1e-15f);
        scale[v]  = tanhf(vn) / vn;
    }
    #pragma unroll
    for (int t = 0; t < 4; ++t) {
        #pragma unroll
        for (int v = 0; v < 8; ++v) {
            out[(size_t)(i0 + v + hi8) * OUTC + head * DH + t * 16 + idx]
                = acc[t][v] * scale[v];
        }
    }
}

extern "C" void kernel_launch(void* const* d_in, const int* in_sizes, int n_in,
                              void* d_out, int out_size, void* d_ws, size_t ws_size,
                              hipStream_t stream) {
    const float* x    = (const float*)d_in[0];
    const float* adj  = (const float*)d_in[1];
    const float* W    = (const float*)d_in[2];
    const float* a_l  = (const float*)d_in[3];
    const float* a_r  = (const float*)d_in[4];
    const float* bias = (const float*)d_in[5];
    float* out = (float*)d_out;

    char* w = (char*)d_ws;
    _Float16* xh = (_Float16*)w;  w += (size_t)NN * INC * 2;     // 2 MB
    _Float16* Wh = (_Float16*)w;  w += (size_t)OUTC * INC * 2;   // 128 KB
    _Float16* hT = (_Float16*)w;  w += (size_t)OUTC * NN * 2;    // 2 MB
    float*    el = (float*)w;     w += (size_t)NN * HEADS * 4;   // 64 KB
    float*    er = (float*)w;     w += (size_t)NN * HEADS * 4;   // 64 KB

    logmap_kernel<<<NN / 8, 256, 0, stream>>>(x, xh);
    wcvt_kernel<<<(OUTC * INC) / 256, 256, 0, stream>>>(W, Wh);
    gemm_h_kernel<<<NN / 16, 128, 0, stream>>>(xh, Wh, hT);
    e_kernel<<<NN / 256, 256, 0, stream>>>(hT, a_l, a_r, el, er);
    attn_kernel<<<NN / 16, 128, 0, stream>>>(adj, hT, el, er, bias, out);
}
